// QuantBertSelfAttention_2to12_46815143526851
// MI455X (gfx1250) — 8-bit matrix operand order repaired and checked by reading, compiled, not run on hardware
//
#include <hip/hip_runtime.h>
#include <stdint.h>

#define HDIM 1024
#define NHEAD 16
#define HD 64
#define BATCH 16
#define SEQ 512
#define MTOT (BATCH * SEQ)
#define QMAXF 127.0f

typedef __attribute__((ext_vector_type(8))) int v8i;


__device__ __forceinline__ float sc_from_max(float m) {
    float s = m / QMAXF;
    return s > 1e-8f ? s : 1e-8f;
}

__device__ __forceinline__ void atomic_fmax_pos(float* addr, float v) {
    atomicMax((unsigned int*)addr, __float_as_uint(v));
}

__device__ __forceinline__ int koffA(int j, int g) {
    return ((j >> 2) << 5) + (((j >> 1) & 1) << 4) + ((j & 1) << 2) + (g << 3);
}
__device__ __forceinline__ int koffB(int j, int g) {
    return ((j >> 2) << 5) + (((j >> 1) & 1) << 4) + ((j & 1) << 2) + (g << 3);
}

__device__ __forceinline__ v8i wmma_iu8(v8i a, v8i b, v8i c) {
    return __builtin_amdgcn_wmma_i32_16x16x64_iu8(true, a, true, b, c, false, false);
}


__global__ void k_init(float* red) {
    if (threadIdx.x < 16) red[threadIdx.x] = 0.f;
}

__global__ __launch_bounds__(256) void k_layernorm_absmax(
    const float* __restrict__ x, const float* __restrict__ gamma,
    const float* __restrict__ beta, float* __restrict__ h, float* red) {
    const int row = blockIdx.x;
    const int t = threadIdx.x;
    const float* xr = x + (size_t)row * HDIM;
    float v[4];
    float s = 0.f, s2 = 0.f;
#pragma unroll
    for (int i = 0; i < 4; ++i) {
        v[i] = xr[t * 4 + i];
        s += v[i];
        s2 += v[i] * v[i];
    }
#pragma unroll
    for (int off = 16; off; off >>= 1) {
        s += __shfl_xor(s, off, 32);
        s2 += __shfl_xor(s2, off, 32);
    }
    __shared__ float ws[8], ws2[8];
    const int wid = t >> 5, lane = t & 31;
    if (lane == 0) { ws[wid] = s; ws2[wid] = s2; }
    __syncthreads();
    if (t == 0) {
        float a = 0.f, b = 0.f;
        for (int i = 0; i < 8; ++i) { a += ws[i]; b += ws2[i]; }
        ws[0] = a; ws2[0] = b;
    }
    __syncthreads();
    const float mu = ws[0] * (1.f / HDIM);
    const float var = ws2[0] * (1.f / HDIM) - mu * mu;
    const float rs = rsqrtf(var + 1e-12f);
    float* hr = h + (size_t)row * HDIM;
    float amax = 0.f;
#pragma unroll
    for (int i = 0; i < 4; ++i) {
        const int c = t * 4 + i;
        const float o = (v[i] - mu) * rs * gamma[c] + beta[c];
        hr[c] = o;
        amax = fmaxf(amax, fabsf(o));
    }
#pragma unroll
    for (int off = 16; off; off >>= 1) amax = fmaxf(amax, __shfl_xor(amax, off, 32));
    if (lane == 0) atomic_fmax_pos(&red[0], amax);
}

__global__ void k_absmax(const float* __restrict__ x, int n, float* red, int idx) {
    float amax = 0.f;
    for (int i = blockIdx.x * blockDim.x + threadIdx.x; i < n; i += gridDim.x * blockDim.x)
        amax = fmaxf(amax, fabsf(x[i]));
#pragma unroll
    for (int off = 16; off; off >>= 1) amax = fmaxf(amax, __shfl_xor(amax, off, 32));
    if ((threadIdx.x & 31) == 0) atomic_fmax_pos(&red[idx], amax);
}

__global__ void k_quant(const float* __restrict__ x, int8_t* __restrict__ q, int n,
                        const float* __restrict__ red, int sidx) {
    const float inv_s = 1.f / sc_from_max(red[sidx]);
    const int i = blockIdx.x * blockDim.x + threadIdx.x;
    if (i >= n) return;
    float r = rintf(x[i] * inv_s);
    r = fminf(fmaxf(r, -128.f), 127.f);
    q[i] = (int8_t)r;
}

__global__ void k_quant_v_t(const float* __restrict__ v, int8_t* __restrict__ vt,
                            const float* __restrict__ red, int sidx) {
    const float inv_s = 1.f / sc_from_max(red[sidx]);
    const int i = blockIdx.x * blockDim.x + threadIdx.x;
    const int d = i & (HD - 1);
    const int h = (i >> 6) & (NHEAD - 1);
    const int row = i >> 10;
    const int b = row >> 9;
    const int s = row & (SEQ - 1);
    float r = rintf(v[i] * inv_s);
    r = fminf(fmaxf(r, -128.f), 127.f);
    const int bh = b * NHEAD + h;
    vt[((size_t)(bh * HD + d)) * SEQ + s] = (int8_t)r;
}

__global__ __launch_bounds__(256) void k_gemm_iu8(
    const int8_t* __restrict__ A, const int8_t* __restrict__ W,
    const float* __restrict__ bias, float* __restrict__ out,
    float* red, int sa, int sw, int somax) {
    const int lane = threadIdx.x & 31, g = lane >> 4, ln16 = lane & 15;
    const int tile = blockIdx.x * 8 + (threadIdx.x >> 5);
    const int tm = (tile >> 4) << 4;
    const int tn = (tile & 15) << 6;
    const int8_t* arow = A + (size_t)(tm + ln16) * HDIM;
    const int8_t* brow0 = W + (size_t)(tn + ln16) * HDIM;
    v8i acc[4];
#pragma unroll
    for (int nt = 0; nt < 4; ++nt)
#pragma unroll
        for (int j = 0; j < 8; ++j) acc[nt][j] = 0;
#pragma unroll 1
    for (int kb = 0; kb < HDIM; kb += 64) {
        v8i a;
#pragma unroll
        for (int j = 0; j < 8; ++j) a[j] = *(const int*)(arow + kb + koffA(j, g));
        __builtin_prefetch(arow + kb + 256, 0, 3);
#pragma unroll
        for (int nt = 0; nt < 4; ++nt) {
            const int8_t* brow = brow0 + (size_t)(nt * 16) * HDIM;
            v8i b;
#pragma unroll
            for (int j = 0; j < 8; ++j) b[j] = *(const int*)(brow + kb + koffB(j, g));
            __builtin_prefetch(brow + kb + 256, 0, 3);
            acc[nt] = wmma_iu8(a, b, acc[nt]);
        }
    }
    const float ssc = sc_from_max(red[sa]) * sc_from_max(red[sw]);
    float amax = 0.f;
#pragma unroll
    for (int nt = 0; nt < 4; ++nt) {
        const int col = tn + nt * 16 + ln16;
        const float bv = bias[col];
#pragma unroll
        for (int r = 0; r < 8; ++r) {
            const int rowi = tm + r + 8 * g;
            const float o = (float)acc[nt][r] * ssc + bv;
            out[(size_t)rowi * HDIM + col] = o;
            amax = fmaxf(amax, fabsf(o));
        }
    }
#pragma unroll
    for (int off = 16; off; off >>= 1) amax = fmaxf(amax, __shfl_xor(amax, off, 32));
    if (lane == 0) atomic_fmax_pos(&red[somax], amax);
}

__global__ __launch_bounds__(256) void k_attn_stats(
    const int8_t* __restrict__ qq, const int8_t* __restrict__ kq,
    const float* __restrict__ mask, float* __restrict__ row_m,
    float* __restrict__ row_z, float* red) {
    const int lane = threadIdx.x & 31, g = lane >> 4, ln16 = lane & 15;
    const int wt = blockIdx.x * 8 + (threadIdx.x >> 5);
    const int bh = wt >> 5;
    const int q0 = (wt & 31) << 4;
    const int b = bh >> 4, h = bh & 15;
    const int8_t* qbase = qq + ((size_t)(b * SEQ)) * HDIM + h * HD;
    const int8_t* kbase = kq + ((size_t)(b * SEQ)) * HDIM + h * HD;
    v8i af;
    const int8_t* arow = qbase + (size_t)(q0 + ln16) * HDIM;
#pragma unroll
    for (int j = 0; j < 8; ++j) af[j] = *(const int*)(arow + koffA(j, g));
    const float ssc = sc_from_max(red[4]) * sc_from_max(red[5]) * 0.125f;
    const float* maskb = mask + b * SEQ;
    float m[8], z[8];
#pragma unroll
    for (int r = 0; r < 8; ++r) { m[r] = -3.4e38f; z[r] = 0.f; }
#pragma unroll 1
    for (int n0 = 0; n0 < SEQ; n0 += 16) {
        const int8_t* brow = kbase + (size_t)(n0 + ln16) * HDIM;
        v8i bf, c;
#pragma unroll
        for (int j = 0; j < 8; ++j) { bf[j] = *(const int*)(brow + koffB(j, g)); c[j] = 0; }
        c = wmma_iu8(af, bf, c);
        const float mk = maskb[n0 + ln16];
#pragma unroll
        for (int r = 0; r < 8; ++r) {
            const float sv = (float)c[r] * ssc + mk;
            const float mn = fmaxf(m[r], sv);
            z[r] = z[r] * __expf(m[r] - mn) + __expf(sv - mn);
            m[r] = mn;
        }
    }
#pragma unroll
    for (int off = 1; off < 16; off <<= 1) {
#pragma unroll
        for (int r = 0; r < 8; ++r) {
            const float mo = __shfl_xor(m[r], off, 32);
            const float zo = __shfl_xor(z[r], off, 32);
            const float mn = fmaxf(m[r], mo);
            z[r] = z[r] * __expf(m[r] - mn) + zo * __expf(mo - mn);
            m[r] = mn;
        }
    }
    if (ln16 == 0) {
#pragma unroll
        for (int r = 0; r < 8; ++r) {
            const int row = q0 + r + 8 * g;
            row_m[bh * SEQ + row] = m[r];
            row_z[bh * SEQ + row] = z[r];
            atomic_fmax_pos(&red[7], 1.f / z[r]);
        }
    }
}

__global__ __launch_bounds__(256) void k_attn_pv(
    const int8_t* __restrict__ qq, const int8_t* __restrict__ kq,
    const int8_t* __restrict__ vt, const float* __restrict__ mask,
    const float* __restrict__ row_m, const float* __restrict__ row_z,
    const float* __restrict__ red, float* __restrict__ out) {
    __shared__ int8_t ptile[8][16 * 68];
    const int lane = threadIdx.x & 31, g = lane >> 4, ln16 = lane & 15;
    const int wid = threadIdx.x >> 5;
    const int wt = blockIdx.x * 8 + wid;
    const int bh = wt >> 5;
    const int q0 = (wt & 31) << 4;
    const int b = bh >> 4, h = bh & 15;
    const int8_t* qbase = qq + ((size_t)(b * SEQ)) * HDIM + h * HD;
    const int8_t* kbase = kq + ((size_t)(b * SEQ)) * HDIM + h * HD;
    const int8_t* vbase = vt + (size_t)bh * HD * SEQ;
    v8i af;
    const int8_t* arow = qbase + (size_t)(q0 + ln16) * HDIM;
#pragma unroll
    for (int j = 0; j < 8; ++j) af[j] = *(const int*)(arow + koffA(j, g));
    const float ssc = sc_from_max(red[4]) * sc_from_max(red[5]) * 0.125f;
    const float sp = sc_from_max(red[7]);
    const float inv_sp = 1.f / sp;
    const float* maskb = mask + b * SEQ;
    float rm[8], rzi[8];
#pragma unroll
    for (int r = 0; r < 8; ++r) {
        const int row = q0 + r + 8 * g;
        rm[r] = row_m[bh * SEQ + row];
        rzi[r] = 1.f / row_z[bh * SEQ + row];
    }
    v8i ctx[4];
#pragma unroll
    for (int nt = 0; nt < 4; ++nt)
#pragma unroll
        for (int j = 0; j < 8; ++j) ctx[nt][j] = 0;
    int8_t* lds = &ptile[wid][0];
#pragma unroll 1
    for (int k0 = 0; k0 < SEQ; k0 += 64) {
#pragma unroll
        for (int nt = 0; nt < 4; ++nt) {
            const int n0 = k0 + nt * 16;
            const int8_t* brow = kbase + (size_t)(n0 + ln16) * HDIM;
            v8i bf, c;
#pragma unroll
            for (int j = 0; j < 8; ++j) { bf[j] = *(const int*)(brow + koffB(j, g)); c[j] = 0; }
            c = wmma_iu8(af, bf, c);
            const int kk = nt * 16 + ln16;
            const float mk = maskb[k0 + kk];
#pragma unroll
            for (int r = 0; r < 8; ++r) {
                const float p = __expf((float)c[r] * ssc + mk - rm[r]) * rzi[r];
                const float pr = fminf(rintf(p * inv_sp), 127.f);
                lds[(r + 8 * g) * 68 + kk] = (int8_t)pr;
            }
        }
        __syncthreads();
        v8i pf;
#pragma unroll
        for (int j = 0; j < 8; ++j)
            pf[j] = *(const int*)(lds + ln16 * 68 + koffA(j, g));
#pragma unroll
        for (int nt = 0; nt < 4; ++nt) {
            const int8_t* vrow = vbase + (size_t)(nt * 16 + ln16) * SEQ + k0;
            v8i vf;
#pragma unroll
            for (int j = 0; j < 8; ++j) vf[j] = *(const int*)(vrow + koffB(j, g));
            ctx[nt] = wmma_iu8(pf, vf, ctx[nt]);
        }
        __syncthreads();
    }
    const float so = sp * sc_from_max(red[6]);
#pragma unroll
    for (int nt = 0; nt < 4; ++nt) {
        const int d = nt * 16 + ln16;
#pragma unroll
        for (int r = 0; r < 8; ++r) {
            const int row = q0 + r + 8 * g;
            out[((size_t)(b * SEQ + row)) * HDIM + h * HD + d] = (float)ctx[nt][r] * so;
        }
    }
}


extern "C" void kernel_launch(void* const* d_in, const int* in_sizes, int n_in,
                              void* d_out, int out_size, void* d_ws, size_t ws_size,
                              hipStream_t stream) {
    (void)in_sizes; (void)n_in; (void)out_size; (void)ws_size;
    const float* x     = (const float*)d_in[0];
    const float* mask  = (const float*)d_in[1];
    const float* gamma = (const float*)d_in[2];
    const float* beta  = (const float*)d_in[3];
    const float* wq    = (const float*)d_in[4];
    const float* bq    = (const float*)d_in[5];
    const float* wk    = (const float*)d_in[6];
    const float* bk    = (const float*)d_in[7];
    const float* wv    = (const float*)d_in[8];
    const float* bv    = (const float*)d_in[9];
    float* out = (float*)d_out;

    const size_t NEL = (size_t)MTOT * HDIM;
    const size_t WEL = (size_t)HDIM * HDIM;
    char* p = (char*)d_ws;
    float*  red   = (float*)p;  p += 256;
    float*  h_f   = (float*)p;  p += NEL * 4;
    float*  q_f   = (float*)p;  p += NEL * 4;
    float*  k_f   = (float*)p;  p += NEL * 4;
    float*  v_f   = (float*)p;  p += NEL * 4;
    int8_t* ha    = (int8_t*)p; p += NEL;
    int8_t* wqq   = (int8_t*)p; p += WEL;
    int8_t* wkq   = (int8_t*)p; p += WEL;
    int8_t* wvq   = (int8_t*)p; p += WEL;
    int8_t* qi    = (int8_t*)p; p += NEL;
    int8_t* ki    = (int8_t*)p; p += NEL;
    int8_t* vti   = (int8_t*)p; p += NEL;
    float*  row_m = (float*)p;  p += (size_t)BATCH * NHEAD * SEQ * 4;
    float*  row_z = (float*)p;  p += (size_t)BATCH * NHEAD * SEQ * 4;

    k_init<<<1, 64, 0, stream>>>(red);
    k_layernorm_absmax<<<MTOT, 256, 0, stream>>>(x, gamma, beta, h_f, red);
    k_absmax<<<1024, 256, 0, stream>>>(wq, (int)WEL, red, 1);
    k_absmax<<<1024, 256, 0, stream>>>(wk, (int)WEL, red, 2);
    k_absmax<<<1024, 256, 0, stream>>>(wv, (int)WEL, red, 3);

    k_quant<<<(int)(NEL / 256), 256, 0, stream>>>(h_f, ha, (int)NEL, red, 0);
    k_quant<<<(int)(WEL / 256), 256, 0, stream>>>(wq, wqq, (int)WEL, red, 1);
    k_quant<<<(int)(WEL / 256), 256, 0, stream>>>(wk, wkq, (int)WEL, red, 2);
    k_quant<<<(int)(WEL / 256), 256, 0, stream>>>(wv, wvq, (int)WEL, red, 3);

    k_gemm_iu8<<<1024, 256, 0, stream>>>(ha, wqq, bq, q_f, red, 0, 1, 4);
    k_gemm_iu8<<<1024, 256, 0, stream>>>(ha, wkq, bk, k_f, red, 0, 2, 5);
    k_gemm_iu8<<<1024, 256, 0, stream>>>(ha, wvq, bv, v_f, red, 0, 3, 6);

    k_quant<<<(int)(NEL / 256), 256, 0, stream>>>(q_f, qi, (int)NEL, red, 4);
    k_quant<<<(int)(NEL / 256), 256, 0, stream>>>(k_f, ki, (int)NEL, red, 5);
    k_quant_v_t<<<(int)(NEL / 256), 256, 0, stream>>>(v_f, vti, red, 6);

    k_attn_stats<<<1024, 256, 0, stream>>>(qi, ki, mask, row_m, row_z, red);
    k_attn_pv<<<1024, 256, 0, stream>>>(qi, ki, vti, mask, row_m, row_z, red, out);
}
